// LaCTBlock_80711025426721
// MI455X (gfx1250) — matrix operand order repaired and checked by reading, not compiled, not run on hardware
//
// MI455X (gfx1250, CDNA5) implementation of the LaCT block.
// Compute-bound (~2 TFLOP vs <0.5 GB traffic, AI > 4000 FLOP/B at 23.3 TB/s)
// -> all heavy math through v_wmma_f32_16x16x32_bf16 (bf16 ops, f32 accum).
// GEMM: 128x128 block tile, 2x2 register tiles/wave = 4 WMMAs per LDS stage.
// Attention: K/V block tiles staged via global_load_async_to_lds_b128 (+
// s_wait_asynccnt), fragments then read from LDS.
#include <hip/hip_runtime.h>
#include <cmath>

#define TSEQ   2048
#define DMODEL 2048
#define NHEADS 16
#define NKVH   4
#define HDIM   128
#define FHID   2048
#define CHUNKT 1024
#define WIN    1024
#define DFFN   8192

typedef __attribute__((ext_vector_type(16))) __bf16 v16bf;
typedef __attribute__((ext_vector_type(8)))  float  v8f;

union FragB { v16bf v; unsigned u[8]; };

__device__ __forceinline__ unsigned short f2bf(float f) {
  union { float f; unsigned u; } c; c.f = f;
  unsigned u = c.u + 0x7FFFu + ((c.u >> 16) & 1u);   // round-to-nearest-even
  return (unsigned short)(u >> 16);
}
__device__ __forceinline__ float sigm_(float x) { return 1.0f / (1.0f + __expf(-x)); }
__device__ __forceinline__ float gelu_(float v) {
  float t = tanhf(0.7978845608028654f * (v + 0.044715f * v * v * v));
  return 0.5f * v * (1.0f + t);
}
// A-fragment (16x32 bf16): lane holds row M=lane&15; K position by VGPR+lane-half.
__device__ __forceinline__ int kA_of(int v, int h) { return ((v < 4) ? 0 : 16) + h * 8 + (v & 3) * 2; }
// B-fragment (32x16 bf16): lane holds col N=lane&15; K position by VGPR+lane-half (same as A).
__device__ __forceinline__ int kB_of(int v, int h) { return ((v < 4) ? 0 : 16) + h * 8 + (v & 3) * 2; }

// Async copy of 16B from global to LDS (ASYNCcnt path).
__device__ __forceinline__ void async_cp16(unsigned lds_addr, unsigned long long gaddr) {
  asm volatile("global_load_async_to_lds_b128 %0, %1, off"
               :: "v"(lds_addr), "v"(gaddr) : "memory");
}
__device__ __forceinline__ void async_wait0() {
  asm volatile("s_wait_asynccnt 0x0" ::: "memory");
}
__device__ __forceinline__ unsigned lds_off(const void* p) {
  return (unsigned)(unsigned long long)p;   // LDS aperture: low 32 bits = LDS addr
}

// ---------------------------------------------------------------------------
// Generic WMMA GEMM:  C = act(alpha*op(A)*op(B) + bias) + beta*Cin
//   TA=0: A is [M,K];  TA=1: A is [K,M]   (TN gemms)
//   TB=0: B is [K,N];  TB=1: B is [N,K]   (NT gemms, i.e. A*B^T)
// 128x128 block tile, 512 threads = 16 waves in 4x4; each wave owns a 32x32
// region = 2x2 WMMA tiles -> 4 v_wmma per K-step from 2 A-frags x 2 B-frags.
// fp32 -> bf16 conversion happens while staging tiles into LDS.
// ---------------------------------------------------------------------------
template<int TA, int TB, int ACT>
__global__ __launch_bounds__(512) void gemm_wmma(
    const float* __restrict__ A, const float* __restrict__ B,
    const float* __restrict__ bias, const float* __restrict__ Cin,
    float* __restrict__ C, int M, int N, int K, float alpha, float beta)
{
  __shared__ __align__(16) unsigned short As[128][34];   // [m][k], 68B rows (4B aligned)
  __shared__ __align__(16) unsigned short Bs[128][34];   // [n][k]
  const int tid  = threadIdx.x;
  const int bm   = blockIdx.y * 128, bn = blockIdx.x * 128;
  const int wave = tid >> 5, lane = tid & 31, lh = lane >> 4, lr = lane & 15;
  const int wm   = (wave >> 2) << 5, wn = (wave & 3) << 5;
  v8f acc00 = {}, acc01 = {}, acc10 = {}, acc11 = {};
  for (int k0 = 0; k0 < K; k0 += 32) {
#pragma unroll
    for (int r = 0; r < 8; r++) {           // 128x32 A tile, 8 elems/thread
      int li = tid + r * 512;
      int m, kk;
      if (TA == 0) { m = li >> 5; kk = li & 31; }     // coalesced over k
      else         { kk = li >> 7; m = li & 127; }    // coalesced over m
      float f = (TA == 0) ? A[(size_t)(bm + m) * K + (k0 + kk)]
                          : A[(size_t)(k0 + kk) * M + (bm + m)];
      As[m][kk] = f2bf(f);
    }
#pragma unroll
    for (int r = 0; r < 8; r++) {           // 128x32 B tile
      int li = tid + r * 512;
      int n, kk;
      if (TB == 0) { kk = li >> 7; n = li & 127; }
      else         { n = li >> 5; kk = li & 31; }
      float f = (TB == 0) ? B[(size_t)(k0 + kk) * N + (bn + n)]
                          : B[(size_t)(bn + n) * K + (k0 + kk)];
      Bs[n][kk] = f2bf(f);
    }
    if (k0 + 32 < K) {  // hint next tile toward L2 (global_prefetch_b8)
      __builtin_prefetch(TA == 0 ? &A[(size_t)(bm + (tid >> 2)) * K + k0 + 32]
                                 : &A[(size_t)(k0 + 32 + (tid & 31)) * M + bm], 0, 1);
    }
    __syncthreads();
    FragB a0, a1, b0, b1;
#pragma unroll
    for (int v = 0; v < 8; v++) {
      int ka = kA_of(v, lh), kb = kB_of(v, lh);
      a0.u[v] = *(const unsigned*)&As[wm + lr][ka];
      a1.u[v] = *(const unsigned*)&As[wm + 16 + lr][ka];
      b0.u[v] = *(const unsigned*)&Bs[wn + lr][kb];
      b1.u[v] = *(const unsigned*)&Bs[wn + 16 + lr][kb];
    }
    acc00 = __builtin_amdgcn_wmma_f32_16x16x32_bf16(false, a0.v, false, b0.v, (short)0, acc00, false, false);
    acc01 = __builtin_amdgcn_wmma_f32_16x16x32_bf16(false, a0.v, false, b1.v, (short)0, acc01, false, false);
    acc10 = __builtin_amdgcn_wmma_f32_16x16x32_bf16(false, a1.v, false, b0.v, (short)0, acc10, false, false);
    acc11 = __builtin_amdgcn_wmma_f32_16x16x32_bf16(false, a1.v, false, b1.v, (short)0, acc11, false, false);
    __syncthreads();
  }
  const v8f* accs[4] = { &acc00, &acc01, &acc10, &acc11 };
#pragma unroll
  for (int t = 0; t < 4; t++) {
    int ti = t >> 1, tj = t & 1;
#pragma unroll
    for (int r = 0; r < 8; r++) {   // C layout: M = r + 8*half, N = lane&15
      size_t row = (size_t)(bm + wm + ti * 16 + r + lh * 8);
      size_t col = (size_t)(bn + wn + tj * 16 + lr);
      float v = alpha * (*accs[t])[r];
      if (bias) v += bias[col];
      if (ACT == 1) v = gelu_(v);
      if (beta != 0.0f) v += beta * Cin[row * (size_t)N + col];
      C[row * (size_t)N + col] = v;
    }
  }
}

// ---------------------------------------------------------------------------
// Flash-style sliding-window GQA attention. One wave (32 thr) per
// (16-query tile, head). WMMA for Q*K^T and P*V, online softmax in f32.
// q: bf16 [T][NH][HD]; k: bf16 [T][NKV][HD]; vT: bf16 [NKV][HD][T].
// K/V block tiles are staged into LDS with async global->LDS copies.
// ---------------------------------------------------------------------------
__global__ __launch_bounds__(32) void attn_kernel(
    const unsigned short* __restrict__ qbf,
    const unsigned short* __restrict__ kbf,
    const unsigned short* __restrict__ vtbf,
    float* __restrict__ out)
{
  __shared__ __align__(16) unsigned short Ks[32][128];   // [local key][hd], 256B rows
  __shared__ __align__(16) unsigned short Vs[128][32];   // [hd][local key], 64B rows
  __shared__ __align__(16) unsigned short Ps[16][32];
  const int h = blockIdx.y, kvh = h >> 2;
  const int q0 = blockIdx.x * 16;
  const int lane = threadIdx.x, lh = lane >> 4, lr = lane & 15;

  FragB qa[4];
#pragma unroll
  for (int s = 0; s < 4; s++)
#pragma unroll
    for (int v = 0; v < 8; v++) {
      int kb = s * 32 + kA_of(v, lh);
      qa[s].u[v] = *(const unsigned*)&qbf[((size_t)(q0 + lr) * NHEADS + h) * HDIM + kb];
    }
  float m_run[8], l_run[8];
  v8f O[8] = {};
#pragma unroll
  for (int r = 0; r < 8; r++) { m_run[r] = -1e30f; l_run[r] = 0.0f; }

  int kb0 = q0 - WIN + 1; if (kb0 < 0) kb0 = 0; kb0 &= ~31;
  for (int kb = kb0; kb <= q0 + 15; kb += 32) {
    // ---- async-stage K tile (32 keys x 128 hd, bf16) : one row per lane ----
    {
      int key = kb + lane; if (key > TSEQ - 1) key = TSEQ - 1;
      unsigned long long gk = (unsigned long long)(const void*)
          &kbf[((size_t)key * NKVH + kvh) * HDIM];
      unsigned lk = lds_off(&Ks[lane][0]);
#pragma unroll
      for (int c = 0; c < 16; c++) async_cp16(lk + c * 16, gk + c * 16);
    }
    // ---- async-stage V tile (128 hd x 32 keys, bf16) : four rows per lane --
    {
#pragma unroll
      for (int rr = 0; rr < 4; rr++) {
        int hd = lane * 4 + rr;
        unsigned long long gv = (unsigned long long)(const void*)
            &vtbf[((size_t)kvh * HDIM + hd) * TSEQ + kb];
        unsigned lv = lds_off(&Vs[hd][0]);
#pragma unroll
        for (int c = 0; c < 4; c++) async_cp16(lv + c * 16, gv + c * 16);
      }
    }
    async_wait0();
    __syncthreads();

    v8f s0 = {}, s1 = {};
#pragma unroll
    for (int s = 0; s < 4; s++) {
      FragB b0, b1;
#pragma unroll
      for (int v = 0; v < 8; v++) {
        int kk = s * 32 + kB_of(v, lh);
        b0.u[v] = *(const unsigned*)&Ks[lr][kk];
        b1.u[v] = *(const unsigned*)&Ks[16 + lr][kk];
      }
      s0 = __builtin_amdgcn_wmma_f32_16x16x32_bf16(false, qa[s].v, false, b0.v, (short)0, s0, false, false);
      s1 = __builtin_amdgcn_wmma_f32_16x16x32_bf16(false, qa[s].v, false, b1.v, (short)0, s1, false, false);
    }
    const float scale = 0.08838834764831845f;   // 1/sqrt(128)
#pragma unroll
    for (int r = 0; r < 8; r++) {
      int i = q0 + r + lh * 8;
      int j0 = kb + lr, j1 = kb + 16 + lr;
      float v0 = (j0 <= i && j0 > i - WIN) ? s0[r] * scale : -1e30f;
      float v1 = (j1 <= i && j1 > i - WIN) ? s1[r] * scale : -1e30f;
      float mx = fmaxf(v0, v1);
      for (int x = 1; x < 16; x <<= 1) mx = fmaxf(mx, __shfl_xor(mx, x, 32));
      float mn = fmaxf(m_run[r], mx);
      float sc = __expf(m_run[r] - mn);
      float p0 = __expf(v0 - mn), p1 = __expf(v1 - mn);
      float rs = p0 + p1;
      for (int x = 1; x < 16; x <<= 1) rs += __shfl_xor(rs, x, 32);
      l_run[r] = l_run[r] * sc + rs;
      m_run[r] = mn;
#pragma unroll
      for (int nt = 0; nt < 8; nt++) O[nt][r] *= sc;
      Ps[r + lh * 8][lr]      = f2bf(p0);
      Ps[r + lh * 8][16 + lr] = f2bf(p1);
    }
    __syncthreads();                   // C-frag -> A-frag re-layout via LDS
    FragB pa;
#pragma unroll
    for (int v = 0; v < 8; v++) pa.u[v] = *(const unsigned*)&Ps[lr][kA_of(v, lh)];
#pragma unroll
    for (int nt = 0; nt < 8; nt++) {
      FragB vb;
#pragma unroll
      for (int v = 0; v < 8; v++)
        vb.u[v] = *(const unsigned*)&Vs[nt * 16 + lr][kB_of(v, lh)];
      O[nt] = __builtin_amdgcn_wmma_f32_16x16x32_bf16(false, pa.v, false, vb.v, (short)0, O[nt], false, false);
    }
    __syncthreads();
  }
#pragma unroll
  for (int nt = 0; nt < 8; nt++)
#pragma unroll
    for (int r = 0; r < 8; r++) {
      size_t row = (size_t)(q0 + r + lh * 8);
      out[row * (size_t)(NHEADS * HDIM) + h * HDIM + nt * 16 + lr] = O[nt][r] / l_run[r];
    }
}

// ------------------------- small VALU kernels ------------------------------
__global__ __launch_bounds__(256) void ln_kernel(const float* __restrict__ x,
    const float* __restrict__ g, const float* __restrict__ b,
    float* __restrict__ y, int D)
{
  __shared__ float rs[256], rq[256];
  int t = blockIdx.x, tid = threadIdx.x;
  const float* xr = x + (size_t)t * D;
  float s = 0, q = 0;
  for (int d = tid; d < D; d += 256) { float v = xr[d]; s += v; q += v * v; }
  rs[tid] = s; rq[tid] = q; __syncthreads();
  for (int o = 128; o > 0; o >>= 1) {
    if (tid < o) { rs[tid] += rs[tid + o]; rq[tid] += rq[tid + o]; }
    __syncthreads();
  }
  float mean = rs[0] / D;
  float var  = rq[0] / D - mean * mean;
  float inv  = rsqrtf(var + 1e-5f);
  for (int d = tid; d < D; d += 256)
    y[(size_t)t * D + d] = (xr[d] - mean) * inv * g[d] + b[d];
}

__global__ __launch_bounds__(256) void eta_kernel(const float* __restrict__ x,
    const float* __restrict__ w, const float* __restrict__ b,
    float* __restrict__ eta, int D)
{
  __shared__ float rs[256];
  int t = blockIdx.x, tid = threadIdx.x;
  float s = 0;
  for (int d = tid; d < D; d += 256) s += x[(size_t)t * D + d] * w[d];
  rs[tid] = s; __syncthreads();
  for (int o = 128; o > 0; o >>= 1) { if (tid < o) rs[tid] += rs[tid + o]; __syncthreads(); }
  if (tid == 0) eta[t] = sigm_(rs[0] + b[0]);
}

__global__ __launch_bounds__(256) void sqsum_kernel(const float* __restrict__ x,
                                                    float* __restrict__ ss, size_t n)
{
  __shared__ float rs[256];
  size_t i0 = (size_t)blockIdx.x * 256 + threadIdx.x;
  size_t st = (size_t)gridDim.x * 256;
  float s = 0;
  for (size_t i = i0; i < n; i += st) { float v = x[i]; s += v * v; }
  rs[threadIdx.x] = s; __syncthreads();
  for (int o = 128; o > 0; o >>= 1) { if ((int)threadIdx.x < o) rs[threadIdx.x] += rs[threadIdx.x + o]; __syncthreads(); }
  if (threadIdx.x == 0) atomicAdd(ss, rs[0]);
}

#define GSL for (size_t i = (size_t)blockIdx.x * 256 + threadIdx.x; i < n; i += (size_t)gridDim.x * 256)

__global__ void scale_norm_kernel(const float* m, const float* ss, float* x, size_t n) {
  float inv = 1.0f / (sqrtf(*ss) + 1e-7f);
  GSL x[i] = m[i] * inv;
}
__global__ void comb_kernel(const float* A, float* A2, size_t n, float cb, float cc) {
  GSL A2[i] = cb * A[i] + cc * A2[i];
}
__global__ void silu_mul_kernel(const float* h, const float* g, float* z, size_t n) {
  GSL { float x = h[i]; z[i] = x * sigm_(x) * g[i]; }
}
__global__ void dz_kernel(float* dz, const float* eta, size_t n, int rowlen) {
  GSL dz[i] = -dz[i] * eta[i / rowlen];
}
__global__ void rowscale_kernel(const float* v, const float* eta, float* o, size_t n, int rowlen) {
  GSL o[i] = v[i] * eta[i / rowlen];
}
__global__ void t1_kernel(const float* dz, const float* hk, const float* gk, float* t, size_t n) {
  GSL { float s = sigm_(hk[i]); t[i] = dz[i] * gk[i] * (s * (1.0f + hk[i] * (1.0f - s))); }
}
__global__ void t3_kernel(const float* dz, const float* hk, float* t, size_t n) {
  GSL { float s = sigm_(hk[i]); t[i] = dz[i] * hk[i] * s; }
}
__global__ void momadd_kernel(float* M, const float* G, size_t n, float mom) {
  GSL M[i] = mom * M[i] + G[i];
}
__global__ void sub_kernel(float* W, const float* X, size_t n) { GSL W[i] -= X[i]; }
__global__ void cvt_bf16_kernel(const float* a, unsigned short* b, size_t n) { GSL b[i] = f2bf(a[i]); }
__global__ void vtrans_kernel(const float* v, unsigned short* o, size_t n) {
  GSL {
    size_t t = i / (NKVH * HDIM);
    size_t r = i % (NKVH * HDIM);
    size_t kvh = r / HDIM, hd = r % HDIM;
    o[(kvh * HDIM + hd) * TSEQ + t] = f2bf(v[i]);
  }
}

// ---------------------------------------------------------------------------
static inline dim3 gblk(size_t n) {
  size_t b = (n + 255) / 256; if (b > 4096) b = 4096;
  return dim3((unsigned)b);
}

extern "C" void kernel_launch(void* const* d_in, const int* in_sizes, int n_in,
                              void* d_out, int out_size, void* d_ws, size_t ws_size,
                              hipStream_t stream)
{
  (void)in_sizes; (void)n_in; (void)out_size; (void)ws_size;
  const float* x       = (const float*)d_in[0];
  const float* ln_a_g  = (const float*)d_in[1];
  const float* ln_a_b  = (const float*)d_in[2];
  const float* ln_b_g  = (const float*)d_in[3];
  const float* ln_b_b  = (const float*)d_in[4];
  const float* ln_c_g  = (const float*)d_in[5];
  const float* ln_c_b  = (const float*)d_in[6];
  const float* attn_wq = (const float*)d_in[7];
  const float* attn_bq = (const float*)d_in[8];
  const float* attn_wk = (const float*)d_in[9];
  const float* attn_bk = (const float*)d_in[10];
  const float* attn_wv = (const float*)d_in[11];
  const float* attn_bv = (const float*)d_in[12];
  const float* attn_wo = (const float*)d_in[13];
  const float* attn_bo = (const float*)d_in[14];
  const float* q_w     = (const float*)d_in[15];
  const float* q_b     = (const float*)d_in[16];
  const float* k_w     = (const float*)d_in[17];
  const float* k_b     = (const float*)d_in[18];
  const float* v_w     = (const float*)d_in[19];
  const float* v_b     = (const float*)d_in[20];
  const float* eta_w   = (const float*)d_in[21];
  const float* eta_b   = (const float*)d_in[22];
  const float* W1_0    = (const float*)d_in[23];
  const float* W2_0    = (const float*)d_in[24];
  const float* W3_0    = (const float*)d_in[25];
  const float* out_w   = (const float*)d_in[26];
  const float* out_b   = (const float*)d_in[27];
  const float* mlp_w1  = (const float*)d_in[28];
  const float* mlp_b1  = (const float*)d_in[29];
  const float* mlp_w2  = (const float*)d_in[30];
  const float* mlp_b2  = (const float*)d_in[31];
  float* xo = (float*)d_out;

  const int T_ = TSEQ, D_ = DMODEL, FH_ = FHID, C_ = CHUNKT, DF_ = DFFN;
  const size_t TD = (size_t)T_ * D_;
  const size_t CF = (size_t)C_ * FH_;
  const size_t TKV = (size_t)T_ * NKVH * HDIM;

  char* base = (char*)d_ws; size_t off = 0;
  auto alloc = [&](size_t bytes) -> void* {
    void* p = base + off; off += (bytes + 255) & ~(size_t)255; return p;
  };
  float* lnb  = (float*)alloc(TD * 4);
  float* qf   = (float*)alloc(TD * 4);
  float* kf   = (float*)alloc(TKV * 4);
  float* vf   = (float*)alloc(TKV * 4);
  unsigned short* qbf = (unsigned short*)alloc(TD * 2);
  unsigned short* kbf = (unsigned short*)alloc(TKV * 2);
  unsigned short* vtb = (unsigned short*)alloc(TKV * 2);
  float* aout = (float*)alloc(TD * 4);
  float* Qb   = (float*)alloc(TD * 4);
  float* Kb   = (float*)alloc(TD * 4);
  float* Vb   = (float*)alloc(TD * 4);
  float* etab = (float*)alloc((size_t)T_ * 4);
  float* hq   = (float*)alloc(CF * 4);
  float* gq   = (float*)alloc(CF * 4);
  float* zz   = (float*)alloc(CF * 4);
  float* hk   = (float*)alloc(CF * 4);
  float* gk   = (float*)alloc(CF * 4);
  float* zk   = (float*)alloc(CF * 4);
  float* dzb  = (float*)alloc(CF * 4);
  float* tb   = (float*)alloc(CF * 4);
  float* veb  = (float*)alloc((size_t)C_ * D_ * 4);
  float* W1   = (float*)alloc(TD * 4);
  float* W2   = (float*)alloc(TD * 4);
  float* W3   = (float*)alloc(TD * 4);
  float* M1   = (float*)alloc(TD * 4);
  float* M2   = (float*)alloc(TD * 4);
  float* M3   = (float*)alloc(TD * 4);
  float* Gb   = (float*)alloc(TD * 4);
  float* Xa   = (float*)alloc(TD * 4);
  float* Xb   = (float*)alloc(TD * 4);
  float* Am   = (float*)alloc(TD * 4);
  float* A2m  = (float*)alloc(TD * 4);
  float* tout = (float*)alloc(TD * 4);
  float* mlph = (float*)alloc((size_t)T_ * DF_ * 4);
  float* ssb  = (float*)alloc(256);

  auto NN = [&](const float* A, const float* B, const float* bias, const float* Cin,
                float* C, int M, int N, int K, float alpha, float beta) {
    gemm_wmma<0, 0, 0><<<dim3(N / 128, M / 128), 512, 0, stream>>>(A, B, bias, Cin, C, M, N, K, alpha, beta);
  };
  auto NNG = [&](const float* A, const float* B, const float* bias, float* C,
                 int M, int N, int K) {
    gemm_wmma<0, 0, 1><<<dim3(N / 128, M / 128), 512, 0, stream>>>(A, B, bias, nullptr, C, M, N, K, 1.0f, 0.0f);
  };
  auto NT = [&](const float* A, const float* B, const float* Cin, float* C,
                int M, int N, int K, float alpha, float beta) {
    gemm_wmma<0, 1, 0><<<dim3(N / 128, M / 128), 512, 0, stream>>>(A, B, nullptr, Cin, C, M, N, K, alpha, beta);
  };
  auto TN = [&](const float* A, const float* B, float* C, int M, int N, int K, float alpha) {
    gemm_wmma<1, 0, 0><<<dim3(N / 128, M / 128), 512, 0, stream>>>(A, B, nullptr, nullptr, C, M, N, K, alpha, 0.0f);
  };

  // residual stream lives in d_out
  hipMemcpyAsync(xo, x, TD * 4, hipMemcpyDeviceToDevice, stream);

  // ---- attention ----
  ln_kernel<<<T_, 256, 0, stream>>>(xo, ln_a_g, ln_a_b, lnb, D_);
  NN(lnb, attn_wq, attn_bq, nullptr, qf, T_, NHEADS * HDIM, D_, 1.0f, 0.0f);
  NN(lnb, attn_wk, attn_bk, nullptr, kf, T_, NKVH * HDIM, D_, 1.0f, 0.0f);
  NN(lnb, attn_wv, attn_bv, nullptr, vf, T_, NKVH * HDIM, D_, 1.0f, 0.0f);
  cvt_bf16_kernel<<<gblk(TD), 256, 0, stream>>>(qf, qbf, TD);
  cvt_bf16_kernel<<<gblk(TKV), 256, 0, stream>>>(kf, kbf, TKV);
  vtrans_kernel<<<gblk(TKV), 256, 0, stream>>>(vf, vtb, TKV);
  attn_kernel<<<dim3(T_ / 16, NHEADS), 32, 0, stream>>>(qbf, kbf, vtb, aout);
  NN(aout, attn_wo, attn_bo, xo, xo, T_, D_, NHEADS * HDIM, 1.0f, 1.0f);

  // ---- TTT fast-weight layer ----
  ln_kernel<<<T_, 256, 0, stream>>>(xo, ln_b_g, ln_b_b, lnb, D_);
  NN(lnb, q_w, q_b, nullptr, Qb, T_, D_, D_, 1.0f, 0.0f);
  NN(lnb, k_w, k_b, nullptr, Kb, T_, D_, D_, 1.0f, 0.0f);
  NN(lnb, v_w, v_b, nullptr, Vb, T_, D_, D_, 1.0f, 0.0f);
  eta_kernel<<<T_, 256, 0, stream>>>(lnb, eta_w, eta_b, etab, D_);
  hipMemcpyAsync(W1, W1_0, TD * 4, hipMemcpyDeviceToDevice, stream);
  hipMemcpyAsync(W2, W2_0, TD * 4, hipMemcpyDeviceToDevice, stream);
  hipMemcpyAsync(W3, W3_0, TD * 4, hipMemcpyDeviceToDevice, stream);
  hipMemsetAsync(M1, 0, TD * 4, stream);
  hipMemsetAsync(M2, 0, TD * 4, stream);
  hipMemsetAsync(M3, 0, TD * 4, stream);

  for (int c = 0; c < 2; c++) {
    const float* qc = Qb + (size_t)c * C_ * D_;
    const float* kc = Kb + (size_t)c * C_ * D_;
    const float* vc = Vb + (size_t)c * C_ * D_;
    const float* ec = etab + (size_t)c * C_;
    float* toc = tout + (size_t)c * C_ * D_;

    // apply with pre-update weights: o = W2^T-gemm of silu(W1 q) * (W3 q)
    NT(qc, W1, nullptr, hq, C_, FH_, D_, 1.0f, 0.0f);
    NT(qc, W3, nullptr, gq, C_, FH_, D_, 1.0f, 0.0f);
    silu_mul_kernel<<<gblk(CF), 256, 0, stream>>>(hq, gq, zz, CF);
    NT(zz, W2, nullptr, toc, C_, D_, FH_, 1.0f, 0.0f);
    // gradients
    NT(kc, W1, nullptr, hk, C_, FH_, D_, 1.0f, 0.0f);
    NT(kc, W3, nullptr, gk, C_, FH_, D_, 1.0f, 0.0f);
    silu_mul_kernel<<<gblk(CF), 256, 0, stream>>>(hk, gk, zk, CF);
    NN(vc, W2, nullptr, nullptr, dzb, C_, FH_, D_, 1.0f, 0.0f);
    dz_kernel<<<gblk(CF), 256, 0, stream>>>(dzb, ec, CF, FH_);
    rowscale_kernel<<<gblk((size_t)C_ * D_), 256, 0, stream>>>(vc, ec, veb, (size_t)C_ * D_, D_);
    TN(veb, zk, Gb, D_, FH_, C_, -1.0f);                       // G2
    momadd_kernel<<<gblk(TD), 256, 0, stream>>>(M2, Gb, TD, 0.9f);
    t1_kernel<<<gblk(CF), 256, 0, stream>>>(dzb, hk, gk, tb, CF);
    TN(tb, kc, Gb, FH_, D_, C_, 1.0f);                         // G1
    momadd_kernel<<<gblk(TD), 256, 0, stream>>>(M1, Gb, TD, 0.9f);
    t3_kernel<<<gblk(CF), 256, 0, stream>>>(dzb, hk, tb, CF);
    TN(tb, kc, Gb, FH_, D_, C_, 1.0f);                         // G3
    momadd_kernel<<<gblk(TD), 256, 0, stream>>>(M3, Gb, TD, 0.9f);

    // Muon (Newton-Schulz, 5 iters) on each momentum; W -= muon(M)
    float* Wm[3] = { W1, W2, W3 };
    float* Mm[3] = { M1, M2, M3 };
    for (int m = 0; m < 3; m++) {
      hipMemsetAsync(ssb, 0, 4, stream);
      sqsum_kernel<<<1024, 256, 0, stream>>>(Mm[m], ssb, TD);
      scale_norm_kernel<<<gblk(TD), 256, 0, stream>>>(Mm[m], ssb, Xa, TD);
      float* Xc = Xa; float* Xn = Xb;
      for (int it = 0; it < 5; it++) {
        NT(Xc, Xc, nullptr, Am, 2048, 2048, 2048, 1.0f, 0.0f);           // A = X X^T
        NN(Am, Am, nullptr, nullptr, A2m, 2048, 2048, 2048, 1.0f, 0.0f); // A2 = A A
        comb_kernel<<<gblk(TD), 256, 0, stream>>>(Am, A2m, TD, -4.7750f, 2.0315f);
        NN(A2m, Xc, nullptr, Xc, Xn, 2048, 2048, 2048, 1.0f, 3.4445f);   // X = aX + (bA+cA2)X
        float* t_ = Xc; Xc = Xn; Xn = t_;
      }
      sub_kernel<<<gblk(TD), 256, 0, stream>>>(Wm[m], Xc, TD);
    }
  }
  NN(tout, out_w, out_b, xo, xo, T_, D_, D_, 1.0f, 1.0f);

  // ---- MLP ----
  ln_kernel<<<T_, 256, 0, stream>>>(xo, ln_c_g, ln_c_b, lnb, D_);
  NNG(lnb, mlp_w1, mlp_b1, mlph, T_, DF_, D_);
  NN(mlph, mlp_w2, mlp_b2, xo, xo, T_, D_, DF_, 1.0f, 1.0f);
}
